// PointNetPPVonMises_55456617726253
// MI455X (gfx1250) — matrix operand order repaired and checked by reading, not compiled, not run on hardware
//
#include <hip/hip_runtime.h>
#include <hip/hip_bf16.h>
#include <math.h>

typedef __attribute__((ext_vector_type(16))) _Float16 v16h;
typedef __attribute__((ext_vector_type(8)))  float    v8f;

#define BDIM 256

// =====================================================================
// Packed fragment layouts (CDNA5 WMMA 16x16x32 f16, wave32):
//  A tile (16 rows m x 32 k), element (m,k):
//     lane = m + 16*((k&8)!=0) ; slot = ((k&16)?8:0) + (k&7)   (16 halfs/lane)
//  B tile (32 k x 16 n), element (k,n):
//     lane = (n&15) + 16*((k&8)!=0) ; slot = ((k&16)?8:0) + (k&7)
//  C/D f32 tile: reg r, lane l -> m = r + 8*(l>=16), n = l&15
// =====================================================================

// --------------------------------------------------------------------
// Fold BN into weights and pack into per-lane WMMA-B fragment layout.
// --------------------------------------------------------------------
__global__ void __launch_bounds__(BDIM) pack_weight_kernel(
    const float* __restrict__ W, const float* __restrict__ b,
    const float* __restrict__ g, const float* __restrict__ be,
    const float* __restrict__ m, const float* __restrict__ v,
    int cin, int cout, _Float16* __restrict__ Wp, float* __restrict__ biasp,
    int has_bn)
{
    const int kchunks = (cin + 31) >> 5;
    const int ntiles  = (cout + 15) >> 4;
    const int K32 = kchunks * 32;
    const int total = K32 * ntiles * 16;
    for (int e = blockIdx.x * blockDim.x + threadIdx.x; e < total;
         e += gridDim.x * blockDim.x) {
        const int n = e / K32;
        const int k = e - n * K32;
        float scale = 1.0f;
        if (has_bn && n < cout) scale = g[n] * rsqrtf(v[n] + 1e-5f);
        float val = 0.0f;
        if (k < cin && n < cout) val = W[(size_t)k * cout + n] * scale;
        const int kc = k >> 5, kin = k & 31, nc = n >> 4, nn = n & 15;
        const int lane = nn + ((kin & 8) ? 16 : 0);
        const int slot = ((kin & 16) ? 8 : 0) + (kin & 7);
        Wp[(((size_t)(kc * ntiles + nc)) * 32 + lane) * 16 + slot] = (_Float16)val;
        if (k == 0) {
            float bb = 0.0f;
            if (n < cout) bb = has_bn ? ((b[n] - m[n]) * scale + be[n]) : b[n];
            biasp[n] = bb;
        }
    }
}

// --------------------------------------------------------------------
// Farthest point sampling: one block per batch; xyz+dist cached in LDS.
// --------------------------------------------------------------------
__global__ void __launch_bounds__(BDIM) fps_kernel(
    const float* __restrict__ xyz, int N, int npoint,
    int* __restrict__ fps_idx, float* __restrict__ new_xyz)
{
    extern __shared__ __align__(16) char smemF[];
    float* pts  = (float*)smemF;   // 3*N
    float* dist = pts + 3 * N;     // N
    __shared__ float sD[BDIM];
    __shared__ int   sI[BDIM];
    const int b = blockIdx.x, tid = threadIdx.x;
    for (int i = tid; i < 3 * N; i += BDIM) pts[i] = xyz[(size_t)b * 3 * N + i];
    for (int i = tid; i < N; i += BDIM) dist[i] = 1e10f;
    __syncthreads();
    int far = 0;
    for (int it = 0; it < npoint; ++it) {
        if (tid == 0) {
            fps_idx[b * npoint + it] = far;
            new_xyz[((size_t)b * npoint + it) * 3 + 0] = pts[far * 3 + 0];
            new_xyz[((size_t)b * npoint + it) * 3 + 1] = pts[far * 3 + 1];
            new_xyz[((size_t)b * npoint + it) * 3 + 2] = pts[far * 3 + 2];
        }
        const float cx = pts[far * 3 + 0], cy = pts[far * 3 + 1], cz = pts[far * 3 + 2];
        float bd = -1.0f; int bi = N;
        for (int i = tid; i < N; i += BDIM) {
            const float dx = pts[i * 3 + 0] - cx;
            const float dy = pts[i * 3 + 1] - cy;
            const float dz = pts[i * 3 + 2] - cz;
            const float d  = dx * dx + dy * dy + dz * dz;
            const float dm = fminf(dist[i], d);
            dist[i] = dm;
            if (dm > bd || (dm == bd && i < bi)) { bd = dm; bi = i; }
        }
        sD[tid] = bd; sI[tid] = bi;
        __syncthreads();
        for (int s = BDIM / 2; s > 0; s >>= 1) {
            if (tid < s) {
                const float od = sD[tid + s]; const int oi = sI[tid + s];
                if (od > sD[tid] || (od == sD[tid] && oi < sI[tid])) { sD[tid] = od; sI[tid] = oi; }
            }
            __syncthreads();
        }
        far = sI[0];
        __syncthreads();
    }
}

// --------------------------------------------------------------------
// kNN (K nearest by squared distance, ties -> lower index).
// --------------------------------------------------------------------
__global__ void __launch_bounds__(BDIM) knn_kernel(
    const float* __restrict__ xyz, const float* __restrict__ centers,
    int N, int S, int K, int* __restrict__ knn_idx)
{
    extern __shared__ __align__(16) char smemK[];
    float* d2 = (float*)smemK;  // N
    __shared__ float sD[BDIM];
    __shared__ int   sI[BDIM];
    const int b = blockIdx.x / S, s = blockIdx.x % S, tid = threadIdx.x;
    const float cx = centers[((size_t)b * S + s) * 3 + 0];
    const float cy = centers[((size_t)b * S + s) * 3 + 1];
    const float cz = centers[((size_t)b * S + s) * 3 + 2];
    for (int i = tid; i < N; i += BDIM) {
        const float dx = xyz[((size_t)b * N + i) * 3 + 0] - cx;
        const float dy = xyz[((size_t)b * N + i) * 3 + 1] - cy;
        const float dz = xyz[((size_t)b * N + i) * 3 + 2] - cz;
        d2[i] = dx * dx + dy * dy + dz * dz;
    }
    __syncthreads();
    for (int j = 0; j < K; ++j) {
        float bd = 3.0e38f; int bi = N;
        for (int i = tid; i < N; i += BDIM) {
            const float d = d2[i];
            if (d < bd || (d == bd && i < bi)) { bd = d; bi = i; }
        }
        sD[tid] = bd; sI[tid] = bi;
        __syncthreads();
        for (int sh = BDIM / 2; sh > 0; sh >>= 1) {
            if (tid < sh) {
                const float od = sD[tid + sh]; const int oi = sI[tid + sh];
                if (od < sD[tid] || (od == sD[tid] && oi < sI[tid])) { sD[tid] = od; sI[tid] = oi; }
            }
            __syncthreads();
        }
        if (tid == 0) { knn_idx[((size_t)(b * S + s)) * K + j] = sI[0]; d2[sI[0]] = 3.4e38f; }
        __syncthreads();
    }
}

// --------------------------------------------------------------------
// One fused GEMM(+bias,+ReLU) layer on WMMA with NB-way register
// blocking over output tiles: one A fragment feeds NB independent WMMA
// chains. WLDS=true: weights were staged to LDS by the caller (ds_load
// path, no prefetch); WLDS=false: weights stream from global with
// unconditional prefetch (buffers padded by one K-step).
// --------------------------------------------------------------------
template <int NB, bool WLDS>
__device__ __forceinline__ void mlp_layer_nb(
    const _Float16* __restrict__ Acur, _Float16* __restrict__ Anext,
    float* __restrict__ outS,
    const _Float16* __restrict__ Wp, const float* __restrict__ bias,
    int tiles, int kchunks, int ntiles, int stageC, bool relu, bool toStage)
{
    const int tid  = threadIdx.x;
    const int lane = tid & 31;
    const int wid  = tid >> 5;
    const int nW   = blockDim.x >> 5;
    const size_t aStride = (size_t)tiles * 512;    // halfs per K-step
    const size_t wStride = (size_t)ntiles * 512;   // halfs per K-step
    for (int t = wid; t < tiles; t += nW) {        // wave-uniform: EXEC all-ones
        for (int nc0 = 0; nc0 < ntiles; nc0 += NB) {
            v8f acc[NB];
#pragma unroll
            for (int j = 0; j < NB; ++j) {
                const float bv = bias[(nc0 + j) * 16 + (lane & 15)];
                v8f tmp = {bv, bv, bv, bv, bv, bv, bv, bv};
                acc[j] = tmp;
            }
            const _Float16* aPtr = Acur + ((size_t)t * 32 + lane) * 16;
            const _Float16* wPtr = Wp + ((size_t)nc0 * 32 + lane) * 16;
            for (int kc = 0; kc < kchunks; ++kc) {
                const v16h a = *(const v16h*)aPtr;
                if (!WLDS)
                    __builtin_prefetch(wPtr + wStride, 0, 1);   // next K-step (padded)
#pragma unroll
                for (int j = 0; j < NB; ++j) {
                    const v16h w = *(const v16h*)(wPtr + (size_t)j * 512);
                    acc[j] = __builtin_amdgcn_wmma_f32_16x16x32_f16(
                        /*neg_a=*/false, a, /*neg_b=*/false, w,
                        /*c_mod=*/(short)0, acc[j], /*reuse_a=*/false, /*reuse_b=*/false);
                }
                aPtr += aStride;
                wPtr += wStride;
            }
            const int n0    = nc0 * 16 + (lane & 15);
            const int mbase = (lane >= 16) ? 8 : 0;
#pragma unroll
            for (int j = 0; j < NB; ++j) {
                const int n = n0 + j * 16;
#pragma unroll
                for (int r = 0; r < 8; ++r) {
                    float x = acc[j][r];
                    if (relu) x = fmaxf(x, 0.0f);
                    const int m = r + mbase;
                    if (toStage) {
                        outS[(size_t)(t * 16 + m) * stageC + n] = x;
                    } else {
                        const int kc2 = n >> 5, kin = n & 31;
                        const int ln = m + ((kin & 8) ? 16 : 0);
                        const int st = ((kin & 16) ? 8 : 0) + (kin & 7);
                        Anext[(((size_t)(kc2 * tiles + t)) * 32 + ln) * 16 + st] = (_Float16)x;
                    }
                }
            }
        }
    }
}

// --------------------------------------------------------------------
// Fused set-abstraction MLP: gather grouped features -> 3x (GEMM+BN+ReLU)
// on WMMA with LDS ping-pong -> max over K neighbors -> new_points.
// Layer-0/1 weights are staged into the (idle until layer 2) outS LDS
// region when they fit, cutting per-block global weight traffic 8x.
// mode 0: kNN grouping (xyz[idx]-center, points[idx]); mode 1: group_all.
// --------------------------------------------------------------------
__global__ void __launch_bounds__(BDIM) sa_mlp_kernel(
    const float* __restrict__ xyz, const float* __restrict__ points,
    const float* __restrict__ centers, const int* __restrict__ knn,
    const _Float16* __restrict__ w0, const float* __restrict__ b0,
    const _Float16* __restrict__ w1, const float* __restrict__ b1,
    const _Float16* __restrict__ w2, const float* __restrict__ b2,
    float* __restrict__ out,
    int N, int S, int K, int Cpts, int C1, int C2, int C3,
    int nblk, int tiles, int mode)
{
    const int b    = blockIdx.x / nblk;
    const int blk  = blockIdx.x % nblk;
    const int rows = tiles * 16;
    const int row0 = blk * rows;
    const int Cin  = 3 + Cpts;
    const int kc0  = (Cin + 31) >> 5;
    const int kc1  = C1 >> 5;
    const int kc2  = C2 >> 5;
    const int nt0  = C1 >> 4, nt1 = C2 >> 4, nt2 = C3 >> 4;
    int maxkc = kc0; if (kc1 > maxkc) maxkc = kc1; if (kc2 > maxkc) maxkc = kc2;

    extern __shared__ __align__(32) char smem[];
    _Float16* A0   = (_Float16*)smem;
    _Float16* A1   = A0 + (size_t)maxkc * tiles * 512;
    float*    outS = (float*)(A1 + (size_t)maxkc * tiles * 512);
    const int tid = threadIdx.x;

    // zero-pad layer-0 A buffer (Cin not a multiple of 32)
    const size_t tot0 = (size_t)kc0 * tiles * 512;
    for (size_t i = tid; i < tot0; i += BDIM) A0[i] = (_Float16)0.0f;
    __syncthreads();

    // gather grouped features, write directly in packed-A layout
    const int totF = rows * Cin;
    for (int e = tid; e < totF; e += BDIM) {
        const int rl = e / Cin;
        const int c  = e - rl * Cin;
        const int grow = row0 + rl;
        float val;
        if (mode == 0) {
            const int s  = grow / K;
            const int kk = grow - s * K;
            const int idx = knn[((size_t)b * S + s) * K + kk];
            if (c < 3) val = xyz[((size_t)b * N + idx) * 3 + c]
                           - centers[((size_t)b * S + s) * 3 + c];
            else       val = points[((size_t)b * N + idx) * Cpts + (c - 3)];
        } else {
            if (c < 3) val = xyz[((size_t)b * N + grow) * 3 + c];
            else       val = points[((size_t)b * N + grow) * Cpts + (c - 3)];
        }
        const int t = rl >> 4, m = rl & 15;
        const int kcc = c >> 5, kin = c & 31;
        const int ln = m + ((kin & 8) ? 16 : 0);
        const int st = ((kin & 16) ? 8 : 0) + (kin & 7);
        A0[(((size_t)(kcc * tiles + t)) * 32 + ln) * 16 + st] = (_Float16)val;
    }
    __syncthreads();

    const size_t outBytes = (size_t)tiles * 16 * C3 * 4;
    _Float16*    wS       = (_Float16*)outS;   // weight staging, idle until layer 2

    // ---- layer 0 ----
    const size_t w0bytes = (size_t)kc0 * nt0 * 1024;
    if (w0bytes <= outBytes) {
        uint4* dst = (uint4*)wS; const uint4* src = (const uint4*)w0;
        for (size_t i = tid; i < (w0bytes >> 4); i += BDIM) dst[i] = src[i];
        __syncthreads();
        mlp_layer_nb<4, true >(A0, A1, nullptr, wS, b0, tiles, kc0, nt0, 0, true, false);
    } else {
        mlp_layer_nb<4, false>(A0, A1, nullptr, w0, b0, tiles, kc0, nt0, 0, true, false);
    }
    __syncthreads();

    // ---- layer 1 ----
    const size_t w1bytes = (size_t)kc1 * nt1 * 1024;
    if (w1bytes <= outBytes) {
        uint4* dst = (uint4*)wS; const uint4* src = (const uint4*)w1;
        for (size_t i = tid; i < (w1bytes >> 4); i += BDIM) dst[i] = src[i];
        __syncthreads();
        mlp_layer_nb<4, true >(A1, A0, nullptr, wS, b1, tiles, kc1, nt1, 0, true, false);
    } else {
        mlp_layer_nb<4, false>(A1, A0, nullptr, w1, b1, tiles, kc1, nt1, 0, true, false);
    }
    __syncthreads();

    // ---- layer 2 (writes outS, weights stream from global) ----
    mlp_layer_nb<4, false>(A0, nullptr, outS, w2, b2, tiles, kc2, nt2, C3, true, true);
    __syncthreads();

    // max-pool over K neighbors
    const int groups = rows / K;
    const int totO = groups * C3;
    for (int e = tid; e < totO; e += BDIM) {
        const int g = e / C3;
        const int c = e - g * C3;
        float mx = -3.0e38f;
        for (int r = 0; r < K; ++r) mx = fmaxf(mx, outS[(size_t)(g * K + r) * C3 + c]);
        const int sIdx = row0 / K + g;
        out[((size_t)b * S + sIdx) * C3 + c] = mx;
    }
}

// --------------------------------------------------------------------
// Head: l3[32,1024] -> 512 -> 256 -> 2 ; mu = tanh*pi, kappa = softplus.
// --------------------------------------------------------------------
__global__ void __launch_bounds__(BDIM) head_kernel(
    const float* __restrict__ l3,
    const _Float16* __restrict__ w0, const float* __restrict__ b0,
    const _Float16* __restrict__ w1, const float* __restrict__ b1,
    const _Float16* __restrict__ w2, const float* __restrict__ b2,
    float* __restrict__ out)
{
    const int tiles = 2;   // 32 rows
    const int kcMax = 32;  // 1024/32
    extern __shared__ __align__(32) char smemH[];
    _Float16* A0   = (_Float16*)smemH;
    _Float16* A1   = A0 + (size_t)kcMax * tiles * 512;
    float*    outS = (float*)(A1 + (size_t)kcMax * tiles * 512);
    const int tid = threadIdx.x;

    for (int e = tid; e < 32 * 1024; e += BDIM) {
        const int rl = e >> 10;
        const int c  = e & 1023;
        const float val = l3[(size_t)rl * 1024 + c];
        const int t = rl >> 4, m = rl & 15;
        const int kcc = c >> 5, kin = c & 31;
        const int ln = m + ((kin & 8) ? 16 : 0);
        const int st = ((kin & 16) ? 8 : 0) + (kin & 7);
        A0[(((size_t)(kcc * tiles + t)) * 32 + ln) * 16 + st] = (_Float16)val;
    }
    __syncthreads();
    mlp_layer_nb<4, false>(A0, A1, nullptr, w0, b0, tiles, 32, 32, 0, true, false);   // 1024->512
    __syncthreads();
    mlp_layer_nb<4, false>(A1, A0, nullptr, w1, b1, tiles, 16, 16, 0, true, false);   // 512->256
    __syncthreads();
    mlp_layer_nb<1, false>(A0, nullptr, outS, w2, b2, tiles, 8, 1, 16, false, true);  // 256->2
    __syncthreads();
    if (tid < 32) {
        const float x0 = outS[tid * 16 + 0];
        const float x1 = outS[tid * 16 + 1];
        out[tid]      = tanhf(x0) * 3.14159265358979323846f;
        out[32 + tid] = (x1 > 20.0f) ? x1 : log1pf(expf(x1));
    }
}

// =====================================================================
// Host launcher
// =====================================================================
extern "C" void kernel_launch(void* const* d_in, const int* in_sizes, int n_in,
                              void* d_out, int out_size, void* d_ws, size_t ws_size,
                              hipStream_t stream)
{
    (void)in_sizes; (void)n_in; (void)out_size; (void)ws_size;
    const int B = 32;
    const float* xyz = (const float*)d_in[0];

    // flat input layout: xyz, 3x(sa1 layer: W,b,g,be,m,v), 3x sa2, 3x sa3,
    // fc1(6), fc2(6), W3, b3  -> 12 GEMM layers total (fc3 has no BN)
    const int base[12]  = {1, 7, 13, 19, 25, 31, 37, 43, 49, 55, 61, 67};
    const int cins[12]  = {3, 64, 64, 131, 128, 128, 259, 256, 512, 1024, 512, 256};
    const int couts[12] = {64, 64, 128, 128, 128, 256, 256, 512, 1024, 512, 256, 2};

    char* ws = (char*)d_ws;
    size_t off = 0;
    auto alloc = [&](size_t bytes) -> void* {
        off = (off + 255) & ~(size_t)255;
        void* p = ws + off;
        off += bytes;
        return p;
    };

    int*   fps1  = (int*)  alloc((size_t)B * 128 * 4);
    float* l1xyz = (float*)alloc((size_t)B * 128 * 3 * 4);
    int*   knn1  = (int*)  alloc((size_t)B * 128 * 32 * 4);
    float* l1pts = (float*)alloc((size_t)B * 128 * 128 * 4);
    int*   fps2  = (int*)  alloc((size_t)B * 32 * 4);
    float* l2xyz = (float*)alloc((size_t)B * 32 * 3 * 4);
    int*   knn2  = (int*)  alloc((size_t)B * 32 * 32 * 4);
    float* l2pts = (float*)alloc((size_t)B * 32 * 256 * 4);
    float* l3    = (float*)alloc((size_t)B * 1024 * 4);

    _Float16* Wp[12];
    float*    Bp[12];
    for (int i = 0; i < 12; ++i) {
        const int kchunks = (cins[i] + 31) / 32;
        const int ntiles  = (couts[i] + 15) / 16;
        const size_t halfs = (size_t)kchunks * ntiles * 512;
        const size_t pad   = (size_t)ntiles * 512;       // one K-step: prefetch overrun
        Wp[i] = (_Float16*)alloc((halfs + pad) * 2);
        Bp[i] = (float*)   alloc((size_t)ntiles * 16 * 4);
        const int has_bn = (i < 11) ? 1 : 0;
        const float* W  = (const float*)d_in[base[i] + 0];
        const float* bb = (const float*)d_in[base[i] + 1];
        const float* g  = has_bn ? (const float*)d_in[base[i] + 2] : nullptr;
        const float* be = has_bn ? (const float*)d_in[base[i] + 3] : nullptr;
        const float* m  = has_bn ? (const float*)d_in[base[i] + 4] : nullptr;
        const float* v  = has_bn ? (const float*)d_in[base[i] + 5] : nullptr;
        int grid = (int)((halfs + BDIM - 1) / BDIM);
        if (grid > 2048) grid = 2048;
        pack_weight_kernel<<<grid, BDIM, 0, stream>>>(W, bb, g, be, m, v,
            cins[i], couts[i], Wp[i], Bp[i], has_bn);
    }

    auto sa_lds = [](int Cin, int C1, int C2, int C3, int tiles) -> size_t {
        const int kc0 = (Cin + 31) / 32, kc1 = C1 / 32, kc2 = C2 / 32;
        int maxkc = kc0; if (kc1 > maxkc) maxkc = kc1; if (kc2 > maxkc) maxkc = kc2;
        return 2 * (size_t)maxkc * tiles * 1024 + (size_t)tiles * 16 * C3 * 4;
    };

    // --- SA1: 8192 -> 128 centers, k=32, MLP 3->64->64->128 ---
    fps_kernel<<<B, BDIM, (size_t)8192 * 16, stream>>>(xyz, 8192, 128, fps1, l1xyz);
    knn_kernel<<<B * 128, BDIM, (size_t)8192 * 4, stream>>>(xyz, l1xyz, 8192, 128, 32, knn1);
    sa_mlp_kernel<<<B * 32, BDIM, sa_lds(3, 64, 64, 128, 8), stream>>>(
        xyz, nullptr, l1xyz, knn1,
        Wp[0], Bp[0], Wp[1], Bp[1], Wp[2], Bp[2], l1pts,
        8192, 128, 32, 0, 64, 64, 128, 32, 8, 0);

    // --- SA2: 128 -> 32 centers, k=32, MLP 131->128->128->256 ---
    fps_kernel<<<B, BDIM, (size_t)128 * 16, stream>>>(l1xyz, 128, 32, fps2, l2xyz);
    knn_kernel<<<B * 32, BDIM, (size_t)128 * 4, stream>>>(l1xyz, l2xyz, 128, 32, 32, knn2);
    sa_mlp_kernel<<<B * 8, BDIM, sa_lds(131, 128, 128, 256, 8), stream>>>(
        l1xyz, l1pts, l2xyz, knn2,
        Wp[3], Bp[3], Wp[4], Bp[4], Wp[5], Bp[5], l2pts,
        128, 32, 32, 128, 128, 128, 256, 8, 8, 0);

    // --- SA3 (group_all): 32 points, MLP 259->256->512->1024 ---
    sa_mlp_kernel<<<B, BDIM, sa_lds(259, 256, 512, 1024, 2), stream>>>(
        l2xyz, l2pts, nullptr, nullptr,
        Wp[6], Bp[6], Wp[7], Bp[7], Wp[8], Bp[8], l3,
        32, 1, 32, 256, 256, 512, 1024, 1, 2, 1);

    // --- Head: 1024 -> 512 -> 256 -> 2 ; (mu, kappa) ---
    const size_t hlds = 2 * (size_t)32 * 2 * 1024 + (size_t)32 * 16 * 4;
    head_kernel<<<1, BDIM, hlds, stream>>>(l3,
        Wp[9], Bp[9], Wp[10], Bp[10], Wp[11], Bp[11], (float*)d_out);
}
